// MultiHeadAttention_28217935135385
// MI455X (gfx1250) — matrix operand order repaired and checked by reading, not compiled, not run on hardware
//
#include <hip/hip_runtime.h>
#include <hip/hip_bf16.h>

typedef __attribute__((ext_vector_type(16))) __bf16 v16bf;
typedef __attribute__((ext_vector_type(8)))  __bf16 v8bf;
typedef __attribute__((ext_vector_type(8)))  float  v8f;

// ---------------------------------------------------------------------------
// Fragment loaders (CDNA5 WMMA bf16 16x16x32 layouts, ISA 7.12.2)
// A 16x32: lane&15 = M row; lanes 0-15 hold K 0-7 (v0-3) & 16-23 (v4-7),
//          lanes 16-31 hold K 8-15 & 24-31.
// B 32x16: lane&15 = N col; same K split as A (lanes 0-15 K 0-7 & 16-23, 16-31 K 8-15 & 24-31).
// `base` points at the first of 32 contiguous K elements for this lane's row.
// ---------------------------------------------------------------------------
__device__ __forceinline__ v16bf load_afrag(const __bf16* base, int lane) {
  const int hb = (lane & 16) ? 8 : 0;
  v8bf a = *(const v8bf*)(base + hb);
  v8bf b = *(const v8bf*)(base + 16 + hb);
  v16bf r;
#pragma unroll
  for (int i = 0; i < 8; ++i) { r[i] = a[i]; r[i + 8] = b[i]; }
  return r;
}

__device__ __forceinline__ v16bf load_bfrag(const __bf16* base, int lane) {
  const int k0 = (lane & 16) ? 8 : 0;
  v8bf a = *(const v8bf*)(base + k0);
  v8bf b = *(const v8bf*)(base + k0 + 16);
  v16bf r;
#pragma unroll
  for (int i = 0; i < 8; ++i) { r[i] = a[i]; r[i + 8] = b[i]; }
  return r;
}

// ---------------------------------------------------------------------------
// Elementwise casts / small prep kernels
// ---------------------------------------------------------------------------
__global__ void cast_f32_bf16_kernel(const float* __restrict__ src,
                                     __bf16* __restrict__ dst, int n) {
  int i = blockIdx.x * blockDim.x + threadIdx.x;
  if (i < n) dst[i] = (__bf16)src[i];
}

// rel_v (33,64) f32 -> rel_v^T (64,64) bf16, zero padded in t (bucket) dim
__global__ void build_relvT_kernel(const float* __restrict__ relv,
                                   __bf16* __restrict__ relvT) {
  int i = blockIdx.x * blockDim.x + threadIdx.x;
  if (i < 64 * 64) {
    int d = i >> 6, t = i & 63;
    float v = (t < 33) ? relv[t * 64 + d] : 0.f;
    relvT[d * 64 + t] = (__bf16)v;
  }
}

// Rq[row, t] = sum_d Qh[row, d] * rel_k[t, d]   (rows = B*H*L, t in [0,33))
__global__ void relq_kernel(const __bf16* __restrict__ Qh,
                            const float* __restrict__ relk,
                            float* __restrict__ Rq, int rows) {
  int idx = blockIdx.x * blockDim.x + threadIdx.x;
  if (idx >= rows * 33) return;
  int row = idx / 33, t = idx % 33;
  const __bf16* q = Qh + (size_t)row * 64;
  const float* rk = relk + t * 64;
  float s = 0.f;
#pragma unroll
  for (int d = 0; d < 64; ++d) s += (float)q[d] * rk[d];
  Rq[idx] = s;
}

// ---------------------------------------------------------------------------
// WMMA GEMM:  Y = X @ W^T + bias.   X:(M,K) bf16 row-major, W:(N,K) bf16.
// One wave computes a 32x64 tile (2 A-frags x 4 B-frags = 8 WMMAs / k-step),
// with explicit software pipelining: k+32 fragments are loaded before the
// current k-step WMMAs so loads overlap the matrix pipe.
// outmode 0: bf16, heads layout (B,H,L,HD)   (for Q,K)
// outmode 1: bf16, transposed heads (B,H,HD,L) (for V^T)
// outmode 2: f32, plain row-major (final output)
// ---------------------------------------------------------------------------
__global__ __launch_bounds__(256) void gemm_wmma(
    const __bf16* __restrict__ X, const __bf16* __restrict__ W,
    const float* __restrict__ bias, void* __restrict__ Y,
    int M, int N, int K, int outmode, int Ldim, int Hdim, int HDdim) {
  const int wave = threadIdx.x >> 5;
  const int lane = threadIdx.x & 31;
  const int gw = blockIdx.x * (blockDim.x >> 5) + wave;
  const int tilesN = N >> 6;
  const int mt = gw / tilesN, nt = gw % tilesN;
  const int m0 = mt * 32, n0 = nt * 64;
  if (m0 >= M) return;

  const int lm = lane & 15;
  v8f acc[2][4] = {};
  const __bf16* arow0 = X + (size_t)(m0 + lm) * K;
  const __bf16* arow1 = X + (size_t)(m0 + 16 + lm) * K;
  const __bf16* wrow0 = W + (size_t)(n0 + 0 * 16 + lm) * K;
  const __bf16* wrow1 = W + (size_t)(n0 + 1 * 16 + lm) * K;
  const __bf16* wrow2 = W + (size_t)(n0 + 2 * 16 + lm) * K;
  const __bf16* wrow3 = W + (size_t)(n0 + 3 * 16 + lm) * K;

  v16bf a0 = load_afrag(arow0, lane);
  v16bf a1 = load_afrag(arow1, lane);
  v16bf b0 = load_bfrag(wrow0, lane);
  v16bf b1 = load_bfrag(wrow1, lane);
  v16bf b2 = load_bfrag(wrow2, lane);
  v16bf b3 = load_bfrag(wrow3, lane);

  for (int k0 = 32; k0 < K; k0 += 32) {
    // issue next k-step fragments before consuming the current ones
    v16bf na0 = load_afrag(arow0 + k0, lane);
    v16bf na1 = load_afrag(arow1 + k0, lane);
    v16bf nb0 = load_bfrag(wrow0 + k0, lane);
    v16bf nb1 = load_bfrag(wrow1 + k0, lane);
    v16bf nb2 = load_bfrag(wrow2 + k0, lane);
    v16bf nb3 = load_bfrag(wrow3 + k0, lane);

    acc[0][0] = __builtin_amdgcn_wmma_f32_16x16x32_bf16(false, a0, false, b0, (short)0, acc[0][0], false, false);
    acc[0][1] = __builtin_amdgcn_wmma_f32_16x16x32_bf16(false, a0, false, b1, (short)0, acc[0][1], false, false);
    acc[0][2] = __builtin_amdgcn_wmma_f32_16x16x32_bf16(false, a0, false, b2, (short)0, acc[0][2], false, false);
    acc[0][3] = __builtin_amdgcn_wmma_f32_16x16x32_bf16(false, a0, false, b3, (short)0, acc[0][3], false, false);
    acc[1][0] = __builtin_amdgcn_wmma_f32_16x16x32_bf16(false, a1, false, b0, (short)0, acc[1][0], false, false);
    acc[1][1] = __builtin_amdgcn_wmma_f32_16x16x32_bf16(false, a1, false, b1, (short)0, acc[1][1], false, false);
    acc[1][2] = __builtin_amdgcn_wmma_f32_16x16x32_bf16(false, a1, false, b2, (short)0, acc[1][2], false, false);
    acc[1][3] = __builtin_amdgcn_wmma_f32_16x16x32_bf16(false, a1, false, b3, (short)0, acc[1][3], false, false);

    a0 = na0; a1 = na1; b0 = nb0; b1 = nb1; b2 = nb2; b3 = nb3;
  }
  // tail k-step
  acc[0][0] = __builtin_amdgcn_wmma_f32_16x16x32_bf16(false, a0, false, b0, (short)0, acc[0][0], false, false);
  acc[0][1] = __builtin_amdgcn_wmma_f32_16x16x32_bf16(false, a0, false, b1, (short)0, acc[0][1], false, false);
  acc[0][2] = __builtin_amdgcn_wmma_f32_16x16x32_bf16(false, a0, false, b2, (short)0, acc[0][2], false, false);
  acc[0][3] = __builtin_amdgcn_wmma_f32_16x16x32_bf16(false, a0, false, b3, (short)0, acc[0][3], false, false);
  acc[1][0] = __builtin_amdgcn_wmma_f32_16x16x32_bf16(false, a1, false, b0, (short)0, acc[1][0], false, false);
  acc[1][1] = __builtin_amdgcn_wmma_f32_16x16x32_bf16(false, a1, false, b1, (short)0, acc[1][1], false, false);
  acc[1][2] = __builtin_amdgcn_wmma_f32_16x16x32_bf16(false, a1, false, b2, (short)0, acc[1][2], false, false);
  acc[1][3] = __builtin_amdgcn_wmma_f32_16x16x32_bf16(false, a1, false, b3, (short)0, acc[1][3], false, false);

  const int hlf = (lane >> 4) & 1;
#pragma unroll
  for (int g = 0; g < 2; ++g) {
#pragma unroll
    for (int f = 0; f < 4; ++f) {
      const int col = n0 + f * 16 + lm;
      const float bval = bias[col];
#pragma unroll
      for (int r = 0; r < 8; ++r) {
        const int mrow = m0 + g * 16 + r + hlf * 8;
        const float val = acc[g][f][r] + bval;
        if (outmode == 2) {
          ((float*)Y)[(size_t)mrow * N + col] = val;
        } else {
          const int bb = mrow / Ldim, l = mrow % Ldim;
          const int h = col >> 6, d = col & 63;
          size_t idx;
          if (outmode == 0)
            idx = (((size_t)(bb * Hdim + h)) * Ldim + l) * HDdim + d;
          else
            idx = (((size_t)(bb * Hdim + h)) * HDdim + d) * Ldim + l;
          ((__bf16*)Y)[idx] = (__bf16)val;
        }
      }
    }
  }
}

// ---------------------------------------------------------------------------
// Fused flash attention with relative position bias (scores) and bucketed
// relative-value accumulation. One wave owns 16 query rows; block = 8 waves
// = 128 query rows of one (b,h). K/V tiles for each 32-key step are staged
// cooperatively into LDS (double-buffered, one step ahead) so the 8 waves
// share one copy instead of issuing 8x redundant global loads; fragments are
// then built with ds_load_b128.
// ---------------------------------------------------------------------------
__global__ __launch_bounds__(256) void attn_kernel(
    const __bf16* __restrict__ Qh, const __bf16* __restrict__ Kh,
    const __bf16* __restrict__ VT, const float* __restrict__ Rq,
    const __bf16* __restrict__ relvT, __bf16* __restrict__ ctx) {
  constexpr int L = 1024, E = 1024;
  constexpr float SCALE = 0.125f;  // 1/sqrt(64)

  __shared__ __align__(16) float  wb[8][16][64];    // bucket mass (padded)
  __shared__ __align__(16) __bf16 pT[8][16][32];    // P tile staging
  __shared__ __align__(16) __bf16 Ktile[2][32][64]; // K rows j..j+31
  __shared__ __align__(16) __bf16 Vtile[2][64][32]; // V^T, d x j
  __shared__ float alphaS[8][16];

  const int wave = threadIdx.x >> 5;
  const int lane = threadIdx.x & 31;
  const int hlf = (lane >> 4) & 1;
  const int lm = lane & 15;
  const int t = threadIdx.x;

  const int qTiles = L / 128;                 // 8
  const int bh = blockIdx.x / qTiles;         // 0..63
  const int qbase = (blockIdx.x % qTiles) * 128 + wave * 16;

  const float* RqBase = Rq + (size_t)bh * L * 33;
  const __bf16* KhBase = Kh + (size_t)bh * L * 64;
  const __bf16* VTBase = VT + (size_t)bh * 64 * L;

  // staging map: thread t moves one 16B chunk of K and one of V per step
  const int krow = t >> 3, kcol = (t & 7) * 8;   // 32 x 64
  const int vrow = t >> 2, vcol = (t & 3) * 8;   // 64 x 32

  for (int idx = lane; idx < 16 * 64; idx += 32)
    wb[wave][idx >> 6][idx & 63] = 0.f;

  // prologue: stage tile j0=0 into buf0, keep tile j0=32 in registers
  v8bf gk = *(const v8bf*)(KhBase + (size_t)krow * 64 + kcol);
  v8bf gv = *(const v8bf*)(VTBase + (size_t)vrow * L + vcol);
  *(v8bf*)&Ktile[0][krow][kcol] = gk;
  *(v8bf*)&Vtile[0][vrow][vcol] = gv;
  gk = *(const v8bf*)(KhBase + (size_t)(32 + krow) * 64 + kcol);
  gv = *(const v8bf*)(VTBase + (size_t)vrow * L + 32 + vcol);

  const __bf16* qrow = Qh + ((size_t)bh * L + qbase + lm) * 64;
  v16bf qa0 = load_afrag(qrow, lane);
  v16bf qa1 = load_afrag(qrow + 32, lane);

  float mrow[8], lrow[8];
#pragma unroll
  for (int r = 0; r < 8; ++r) { mrow[r] = -1e30f; lrow[r] = 0.f; }
  v8f o[4] = {};

  __syncthreads();  // buf0 visible

  for (int j0 = 0; j0 < L; j0 += 32) {
    const int cur = (j0 >> 5) & 1, nxt = cur ^ 1;

    // ---- fragments for this step from LDS ----
    v16bf kf0 = load_bfrag(&Ktile[cur][lm][0], lane);
    v16bf kf1 = load_bfrag(&Ktile[cur][lm][32], lane);
    v16bf kf2 = load_bfrag(&Ktile[cur][16 + lm][0], lane);
    v16bf kf3 = load_bfrag(&Ktile[cur][16 + lm][32], lane);
    v16bf vf0 = load_bfrag(&Vtile[cur][0 * 16 + lm][0], lane);
    v16bf vf1 = load_bfrag(&Vtile[cur][1 * 16 + lm][0], lane);
    v16bf vf2 = load_bfrag(&Vtile[cur][2 * 16 + lm][0], lane);
    v16bf vf3 = load_bfrag(&Vtile[cur][3 * 16 + lm][0], lane);

    // ---- S = (Q K^T + rel) * scale, two 16x16 tiles ----
    v8f s[2];
    {
      v8f z = {};
      z = __builtin_amdgcn_wmma_f32_16x16x32_bf16(false, qa0, false, kf0, (short)0, z, false, false);
      z = __builtin_amdgcn_wmma_f32_16x16x32_bf16(false, qa1, false, kf1, (short)0, z, false, false);
      s[0] = z;
      v8f z2 = {};
      z2 = __builtin_amdgcn_wmma_f32_16x16x32_bf16(false, qa0, false, kf2, (short)0, z2, false, false);
      z2 = __builtin_amdgcn_wmma_f32_16x16x32_bf16(false, qa1, false, kf3, (short)0, z2, false, false);
      s[1] = z2;
    }
#pragma unroll
    for (int jj = 0; jj < 2; ++jj) {
      const int col = j0 + jj * 16 + lm;
#pragma unroll
      for (int r = 0; r < 8; ++r) {
        const int i = qbase + r + hlf * 8;
        int dlt = col - i;
        dlt = dlt < -16 ? -16 : (dlt > 16 ? 16 : dlt);
        s[jj][r] = (s[jj][r] + RqBase[(size_t)i * 33 + dlt + 16]) * SCALE;
      }
    }

    // ---- online softmax: row max across the 16 lanes of this half ----
    float tmax[8], alpha[8];
#pragma unroll
    for (int r = 0; r < 8; ++r) tmax[r] = fmaxf(s[0][r], s[1][r]);
#pragma unroll
    for (int off = 1; off < 16; off <<= 1)
#pragma unroll
      for (int r = 0; r < 8; ++r)
        tmax[r] = fmaxf(tmax[r], __shfl_xor(tmax[r], off, 32));
#pragma unroll
    for (int r = 0; r < 8; ++r) {
      const float mn = fmaxf(mrow[r], tmax[r]);
      alpha[r] = __expf(mrow[r] - mn);
      mrow[r] = mn;
      lrow[r] *= alpha[r];
#pragma unroll
      for (int f = 0; f < 4; ++f) o[f][r] *= alpha[r];
    }
    if (lm == 0) {
#pragma unroll
      for (int r = 0; r < 8; ++r) alphaS[wave][r + hlf * 8] = alpha[r];
    }
    __syncthreads();
    for (int idx = lane; idx < 16 * 33; idx += 32)
      wb[wave][idx / 33][idx % 33] *= alphaS[wave][idx / 33];
    __syncthreads();

    // ---- P = exp(S - m); accumulate row sums and distance buckets ----
    float lsum[8];
#pragma unroll
    for (int r = 0; r < 8; ++r) lsum[r] = 0.f;
#pragma unroll
    for (int jj = 0; jj < 2; ++jj) {
      const int col = j0 + jj * 16 + lm;
#pragma unroll
      for (int r = 0; r < 8; ++r) {
        const int m = r + hlf * 8;
        const int i = qbase + m;
        const float p = __expf(s[jj][r] - mrow[r]);
        lsum[r] += p;
        int dlt = col - i;
        dlt = dlt < -16 ? -16 : (dlt > 16 ? 16 : dlt);
        atomicAdd(&wb[wave][m][dlt + 16], p);
        pT[wave][m][jj * 16 + lm] = (__bf16)p;
      }
    }
#pragma unroll
    for (int off = 1; off < 16; off <<= 1)
#pragma unroll
      for (int r = 0; r < 8; ++r) lsum[r] += __shfl_xor(lsum[r], off, 32);
#pragma unroll
    for (int r = 0; r < 8; ++r) lrow[r] += lsum[r];

    // ---- stage tile j0+32 into LDS; start global loads for j0+64 ----
    if (j0 + 32 < L) {
      *(v8bf*)&Ktile[nxt][krow][kcol] = gk;
      *(v8bf*)&Vtile[nxt][vrow][vcol] = gv;
      if (j0 + 64 < L) {
        gk = *(const v8bf*)(KhBase + (size_t)(j0 + 64 + krow) * 64 + kcol);
        gv = *(const v8bf*)(VTBase + (size_t)vrow * L + j0 + 64 + vcol);
        __builtin_prefetch(KhBase + (size_t)(j0 + 96 + krow) * 64 + kcol, 0, 1);
      }
    }
    __syncthreads();  // fences pT stores and K/V staging stores

    // ---- O += P @ V ----
    v16bf pa = load_afrag(&pT[wave][lm][0], lane);
    o[0] = __builtin_amdgcn_wmma_f32_16x16x32_bf16(false, pa, false, vf0, (short)0, o[0], false, false);
    o[1] = __builtin_amdgcn_wmma_f32_16x16x32_bf16(false, pa, false, vf1, (short)0, o[1], false, false);
    o[2] = __builtin_amdgcn_wmma_f32_16x16x32_bf16(false, pa, false, vf2, (short)0, o[2], false, false);
    o[3] = __builtin_amdgcn_wmma_f32_16x16x32_bf16(false, pa, false, vf3, (short)0, o[3], false, false);
    __syncthreads();  // end-of-step: buf[nxt] ready, buf[cur] free
  }

  // ---- O += Wbuckets @ rel_v  (two K=32 steps over padded 64 buckets) ----
  v16bf wfa[2];
#pragma unroll
  for (int kk = 0; kk < 2; ++kk) {
    const int hb = (lane & 16) ? 8 : 0;
    const int c0 = kk * 32 + hb;
    v16bf w;
#pragma unroll
    for (int i2 = 0; i2 < 8; ++i2) {
      w[i2] = (__bf16)wb[wave][lm][c0 + i2];
      w[i2 + 8] = (__bf16)wb[wave][lm][c0 + 16 + i2];
    }
    wfa[kk] = w;
  }
#pragma unroll
  for (int f = 0; f < 4; ++f) {
#pragma unroll
    for (int kk = 0; kk < 2; ++kk) {
      const __bf16* rrow = relvT + (f * 16 + lm) * 64 + kk * 32;
      v16bf rb = load_bfrag(rrow, lane);
      o[f] = __builtin_amdgcn_wmma_f32_16x16x32_bf16(false, wfa[kk], false, rb,
                                                     (short)0, o[f], false, false);
    }
  }

  // ---- normalize, write ctx in (B, L, E) layout for the output GEMM ----
  const int bb = bh >> 4, h = bh & 15;
#pragma unroll
  for (int r = 0; r < 8; ++r) {
    const int i = qbase + r + hlf * 8;
    const float inv = 1.f / lrow[r];
    const size_t rowoff = ((size_t)bb * L + i) * E + h * 64;
#pragma unroll
    for (int f = 0; f < 4; ++f)
      ctx[rowoff + f * 16 + lm] = (__bf16)(o[f][r] * inv);
  }
}

// ---------------------------------------------------------------------------
// Host launcher
// ---------------------------------------------------------------------------
extern "C" void kernel_launch(void* const* d_in, const int* in_sizes, int n_in,
                              void* d_out, int out_size, void* d_ws, size_t ws_size,
                              hipStream_t stream) {
  (void)in_sizes; (void)n_in; (void)out_size; (void)ws_size;
  constexpr int B = 4, L = 1024, E = 1024, H = 16, HD = 64;

  const float* q    = (const float*)d_in[0];
  const float* k    = (const float*)d_in[1];
  const float* v    = (const float*)d_in[2];
  const float* wq   = (const float*)d_in[3];
  const float* bq   = (const float*)d_in[4];
  const float* wk   = (const float*)d_in[5];
  const float* bk   = (const float*)d_in[6];
  const float* wv   = (const float*)d_in[7];
  const float* bv   = (const float*)d_in[8];
  const float* wo   = (const float*)d_in[9];
  const float* bo   = (const float*)d_in[10];
  const float* relk = (const float*)d_in[11];
  const float* relv = (const float*)d_in[12];
  float* out = (float*)d_out;

  char* ws = (char*)d_ws;
  size_t off = 0;
  auto alloc = [&](size_t bytes) {
    size_t r = off;
    off = (off + bytes + 255) & ~(size_t)255;
    return r;
  };
  const size_t NX = (size_t)B * L * E;  // 4M activations
  const size_t NW = (size_t)E * E;      // 1M weights

  const size_t o_qbf = alloc(NX * 2);   // reused by ctx after GEMM-Q
  const size_t o_kbf = alloc(NX * 2);   // reused (with o_vbf) by Rq after GEMM-V
  const size_t o_vbf = alloc(NX * 2);
  const size_t o_wqb = alloc(NW * 2);
  const size_t o_wkb = alloc(NW * 2);
  const size_t o_wvb = alloc(NW * 2);
  const size_t o_wob = alloc(NW * 2);
  const size_t o_rvt = alloc(64 * 64 * 2);
  const size_t o_Qh  = alloc(NX * 2);
  const size_t o_Kh  = alloc(NX * 2);
  const size_t o_VT  = alloc(NX * 2);
  (void)o_vbf;

  __bf16* qbf = (__bf16*)(ws + o_qbf);
  __bf16* kbf = (__bf16*)(ws + o_kbf);
  __bf16* vbf = (__bf16*)(ws + o_vbf);
  __bf16* wqb = (__bf16*)(ws + o_wqb);
  __bf16* wkb = (__bf16*)(ws + o_wkb);
  __bf16* wvb = (__bf16*)(ws + o_wvb);
  __bf16* wob = (__bf16*)(ws + o_wob);
  __bf16* relvT = (__bf16*)(ws + o_rvt);
  __bf16* Qh = (__bf16*)(ws + o_Qh);
  __bf16* Kh = (__bf16*)(ws + o_Kh);
  __bf16* VT = (__bf16*)(ws + o_VT);
  float*  Rq  = (float*)(ws + o_kbf);   // alias: kbf/vbf dead after GEMM-V
  __bf16* ctx = (__bf16*)(ws + o_qbf);  // alias: qbf dead after GEMM-Q

  auto cast = [&](const float* s, __bf16* d, size_t n) {
    cast_f32_bf16_kernel<<<(unsigned)((n + 255) / 256), 256, 0, stream>>>(s, d, (int)n);
  };
  cast(q, qbf, NX);
  cast(k, kbf, NX);
  cast(v, vbf, NX);
  cast(wq, wqb, NW);
  cast(wk, wkb, NW);
  cast(wv, wvb, NW);
  cast(wo, wob, NW);
  build_relvT_kernel<<<16, 256, 0, stream>>>(relv, relvT);

  const int gemmBlocks = (B * L / 32) * (E / 64) / 8;  // 256
  gemm_wmma<<<gemmBlocks, 256, 0, stream>>>(qbf, wqb, bq, (void*)Qh, B * L, E, E, 0, L, H, HD);
  gemm_wmma<<<gemmBlocks, 256, 0, stream>>>(kbf, wkb, bk, (void*)Kh, B * L, E, E, 0, L, H, HD);
  gemm_wmma<<<gemmBlocks, 256, 0, stream>>>(vbf, wvb, bv, (void*)VT, B * L, E, E, 1, L, H, HD);

  const int rows = B * H * L;  // 65536
  relq_kernel<<<(rows * 33 + 255) / 256, 256, 0, stream>>>(Qh, relk, Rq, rows);

  attn_kernel<<<B * H * (L / 128), 256, 0, stream>>>(Qh, Kh, VT, Rq, relvT, ctx);

  gemm_wmma<<<gemmBlocks, 256, 0, stream>>>(ctx, wob, bo, (void*)out, B * L, E, E, 2, L, H, HD);
}
